// SMAFormerV2_627065225629
// MI455X (gfx1250) — matrix operand order repaired and checked by reading, not compiled, not run on hardware
//
#include <hip/hip_runtime.h>
#include <hip/hip_bf16.h>

typedef unsigned int u32;
typedef unsigned short u16s;

typedef __bf16 v16bf __attribute__((ext_vector_type(16)));
typedef float  v8f   __attribute__((ext_vector_type(8)));
typedef unsigned int v4u __attribute__((ext_vector_type(4)));
typedef int v8i __attribute__((ext_vector_type(8)));
typedef int v4i __attribute__((ext_vector_type(4)));

union Frag { u32 u[8]; u16s h[16]; v16bf v; };

__device__ __forceinline__ u16s f2bf(float f) {
  u32 x = __float_as_uint(f);
  return (u16s)((x + 0x7FFFu + ((x >> 16) & 1u)) >> 16);
}

// ---------------------------------------------------------------- convert
__global__ __launch_bounds__(256) void cvt_bf16_kernel(const float* __restrict__ in,
                                                       u16s* __restrict__ out, int n) {
  int i = blockIdx.x * 256 + threadIdx.x;
  if (i < n) out[i] = f2bf(in[i]);
}

// ---------------------------------------------------------------- LayerNorm (+optional shift/window-partition)
// one wave32 per token; 192 channels -> 6 per lane
__global__ __launch_bounds__(256) void ln_kernel(const float* __restrict__ x,
                                                 const float* __restrict__ g,
                                                 const float* __restrict__ bta,
                                                 u16s* __restrict__ out,
                                                 int shift, int windowed) {
  int token = blockIdx.x * 8 + (threadIdx.x >> 5);
  int lane = threadIdx.x & 31;
  size_t srow;
  if (windowed) {
    int win = token / 49, n = token % 49;
    int bb = win >> 6, wl = win & 63;
    int hr = (wl >> 3) * 7 + n / 7, hc = (wl & 7) * 7 + n % 7;
    int sr = (hr + shift) % 56, sc = (hc + shift) % 56;   // roll by -shift
    srow = (size_t)bb * 3136 + (size_t)sr * 56 + sc;
  } else {
    srow = (size_t)token;
  }
  const float* xp = x + srow * 192;
  float vloc[6];
  float s = 0.f, s2 = 0.f;
#pragma unroll
  for (int j = 0; j < 6; ++j) {
    float v = xp[lane + j * 32];
    vloc[j] = v; s += v; s2 += v * v;
  }
#pragma unroll
  for (int off = 16; off > 0; off >>= 1) {
    s  += __shfl_xor(s,  off, 32);
    s2 += __shfl_xor(s2, off, 32);
  }
  float mean = s * (1.0f / 192.0f);
  float var  = s2 * (1.0f / 192.0f) - mean * mean;
  float rstd = rsqrtf(var + 1e-5f);
  u16s* op = out + (size_t)token * 192;
#pragma unroll
  for (int j = 0; j < 6; ++j) {
    int c = lane + j * 32;
    op[c] = f2bf((vloc[j] - mean) * rstd * g[c] + bta[c]);
  }
}

// ---------------------------------------------------------------- TDM descriptor issue: 64x32 bf16 tile, Global -> LDS
// D# layout per CDNA5 ISA ch.8: group0 {count, lds_addr, global_addr, type=2},
// group1 {data_size=2B, tensor_dim0/1, tile_dim0=32, tile_dim1=64, dim0_stride=K}
__device__ __forceinline__ void tdm_load_tile64x32(const u16s* gptr, u32 lds_off, int K) {
  unsigned long long ga = (unsigned long long)(size_t)gptr;
  v4u g0;
  g0.x = 1u;                                            // count=1, user descriptor
  g0.y = lds_off;                                       // lds_addr (bytes)
  g0.z = (u32)(ga & 0xFFFFFFFFu);                       // global_addr[31:0]
  g0.w = (u32)((ga >> 32) & 0x01FFFFFFu) | (2u << 30);  // global_addr[56:32] | type=2
  v8i g1;
  g1[0] = 0x00010000;                                   // wg_mask=0, data_size=1 (2 bytes)
  g1[1] = (int)(((u32)K & 0xFFFFu) << 16);              // tensor_dim0[15:0]  @ bits[63:48]
  g1[2] = (int)((((u32)K >> 16) & 0xFFFFu)              // tensor_dim0[31:16] @ bits[79:64]
                | (0x4000u << 16));                     // tensor_dim1[15:0]  (large, tiles in-bounds)
  g1[3] = (int)(0u | (32u << 16));                      // tensor_dim1[31:16]=0 | tile_dim0=32
  g1[4] = 64;                                           // tile_dim1=64, tile_dim2=0
  g1[5] = (int)(u32)K;                                  // tensor_dim0_stride[31:0]
  g1[6] = 0;                                            // stride hi | dim1_stride lo
  g1[7] = 0;
  v4i z4 = {0, 0, 0, 0};
  v8i z8 = {0, 0, 0, 0, 0, 0, 0, 0};
  __builtin_amdgcn_tensor_load_to_lds(g0, g1, z4, z4, z8, 0);
}

// ---------------------------------------------------------------- tiled bf16 GEMM, TDM-fed A, fused epilogue
enum { EPI_BF16 = 0, EPI_GELU_BF16 = 1, EPI_F32 = 2, EPI_RESID_F32 = 3 };

__global__ __launch_bounds__(256) void gemm_bf16_kernel(const u16s* __restrict__ A,
                                                        const u16s* __restrict__ W,
                                                        const float* __restrict__ bias,
                                                        void* __restrict__ outp,
                                                        const float* __restrict__ resid,
                                                        int K, int N, int mode) {
  __shared__ u16s As[2][64 * 32];   // [m][k], double buffered, TDM-filled
  __shared__ u16s Bs[2][64 * 32];   // transposed [n][k], double buffered
  const int tid = threadIdx.x;
  const int rowBase = blockIdx.x * 64;
  const int colBase = blockIdx.y * 64;
  const int bk = tid >> 3, bn = (tid & 7) * 8;   // B: 32 rows x 8 x 8-half chunks
  const int lane = tid & 31, wid = tid >> 5;
  const int wm = wid >> 1, wn = wid & 1;
  const int lm = lane & 15, grp = lane >> 4;
  const int nk = K >> 5;

  const u32 as_off0 = (u32)(size_t)&As[0][0];
  const u32 as_off1 = (u32)(size_t)&As[1][0];
  const u16s* Arow = A + (size_t)rowBase * K;

  v8f acc0 = {};
  v8f acc1 = {};

  // prologue: TDM tile 0 in flight, B tile 0 in registers
  if (wid == 0) tdm_load_tile64x32(Arow, as_off0, K);
  uint4 breg = *(const uint4*)(W + (size_t)bk * N + colBase + bn);

  for (int i = 0; i < nk; ++i) {
    const int cur = i & 1;
    // stage current B tile (transposed) into LDS
    u16s tmp[8];
    *(uint4*)tmp = breg;
#pragma unroll
    for (int u = 0; u < 8; ++u) Bs[cur][(bn + u) * 32 + bk] = tmp[u];

    // issue next A tile via TDM into the other buffer (safe: freed by last iteration's end barrier)
    if (wid == 0) {
      if (i + 1 < nk) tdm_load_tile64x32(Arow + (i + 1) * 32, cur ? as_off0 : as_off1, K);
    }
    // fetch next B tile into registers
    if (i + 1 < nk)
      breg = *(const uint4*)(W + (size_t)((i + 1) * 32 + bk) * N + colBase + bn);
    // wait for current A tile (TDM ops retire in order; <=2 in flight)
    if (wid == 0) {
      if (i + 1 < nk) __builtin_amdgcn_s_wait_tensorcnt(1);
      else            __builtin_amdgcn_s_wait_tensorcnt(0);
    }
    __syncthreads();   // publish As[cur] (TDM) and Bs[cur] (ds stores)

    // A fragment: 16x32 bf16, contiguous k order (k = 16*grp + element), same as B
    Frag a;
#pragma unroll
    for (int v = 0; v < 8; ++v) {
      int kb = v * 2 + grp * 16;
      a.u[v] = *(const u32*)&As[cur][(wm * 16 + lm) * 32 + kb];
    }
    // B fragments: 32x16 bf16; contiguous dwords thanks to [n][k] LDS layout
    Frag b0, b1;
#pragma unroll
    for (int v = 0; v < 8; ++v) {
      int kb = v * 2 + grp * 16;
      b0.u[v] = *(const u32*)&Bs[cur][(wn * 32 + lm) * 32 + kb];
      b1.u[v] = *(const u32*)&Bs[cur][(wn * 32 + 16 + lm) * 32 + kb];
    }
    acc0 = __builtin_amdgcn_wmma_f32_16x16x32_bf16(false, a.v, false, b0.v, (short)0, acc0, false, false);
    acc1 = __builtin_amdgcn_wmma_f32_16x16x32_bf16(false, a.v, false, b1.v, (short)0, acc1, false, false);
    __syncthreads();   // everyone done with As[cur]/Bs[cur] before they are refilled
  }

#pragma unroll
  for (int s = 0; s < 2; ++s) {
    v8f acc = s ? acc1 : acc0;
    int col = colBase + wn * 32 + s * 16 + lm;
    float bv = bias[col];
#pragma unroll
    for (int r = 0; r < 8; ++r) {
      int row = rowBase + wm * 16 + grp * 8 + r;
      size_t idx = (size_t)row * N + col;
      float val = acc[r] + bv;
      if (mode == EPI_BF16) {
        ((u16s*)outp)[idx] = f2bf(val);
      } else if (mode == EPI_GELU_BF16) {
        float ge = 0.5f * val * (1.0f + erff(val * 0.70710678118654752f));
        ((u16s*)outp)[idx] = f2bf(ge);
      } else if (mode == EPI_F32) {
        ((float*)outp)[idx] = val;
      } else {
        ((float*)outp)[idx] = val + resid[idx];
      }
    }
  }
}

// ---------------------------------------------------------------- fused windowed attention
// one block per window (4096), 6 waves = 6 heads, 96KB dynamic LDS (6 x 64x64 f32)
__device__ __forceinline__ int reg3(int g) { return g < 49 ? 0 : (g < 53 ? 1 : 2); }

__global__ __launch_bounds__(192) void attn_kernel(const u16s* __restrict__ qkv,
                                                   const float* __restrict__ rpb,
                                                   u16s* __restrict__ out, int shift) {
  extern __shared__ float smem[];
  const int win = blockIdx.x;
  const int head = threadIdx.x >> 5;
  const int lane = threadIdx.x & 31;
  const int lm = lane & 15, grp = lane >> 4;
  float* S = smem + (size_t)head * 64 * 64;
  u16s* P = (u16s*)S;
  const size_t base = (size_t)win * 49 * 576;

  // ---- scores = Q @ K^T (padded 64x64, 16 WMMAs) ----
  Frag bf[4];
#pragma unroll
  for (int ni = 0; ni < 4; ++ni) {
    int key = ni * 16 + lm;
#pragma unroll
    for (int v = 0; v < 8; ++v) {
      bf[ni].u[v] = (key < 49)
          ? *(const u32*)&qkv[base + (size_t)key * 576 + 192 + head * 32 + v * 2 + grp * 16]
          : 0u;
    }
  }
#pragma unroll
  for (int mi = 0; mi < 4; ++mi) {
    int m = mi * 16 + lm;
    Frag a;
#pragma unroll
    for (int v = 0; v < 8; ++v) {
      int kb = v * 2 + grp * 16;
      a.u[v] = (m < 49) ? *(const u32*)&qkv[base + (size_t)m * 576 + head * 32 + kb] : 0u;
    }
#pragma unroll
    for (int ni = 0; ni < 4; ++ni) {
      v8f acc = {};
      acc = __builtin_amdgcn_wmma_f32_16x16x32_bf16(false, a.v, false, bf[ni].v, (short)0, acc, false, false);
#pragma unroll
      for (int r = 0; r < 8; ++r)
        S[(mi * 16 + grp * 8 + r) * 64 + ni * 16 + lm] = acc[r];
    }
  }
  __syncthreads();

  // ---- softmax with rel-pos bias + shift mask; lane handles rows lane, lane+32 ----
  const int wl = win & 63, wr = wl >> 3, wc = wl & 7;
  const float SCALE = 0.17677669529663687f;  // 32^-0.5
#pragma unroll
  for (int rr = 0; rr < 2; ++rr) {
    int i = lane + rr * 32;
    if (i < 49) {
      int ri = i / 7, ci = i % 7;
      int regi = shift ? (reg3(wr * 7 + ri) * 3 + reg3(wc * 7 + ci)) : 0;
      float* Si = S + (size_t)i * 64;
      float mx = -1e30f;
      for (int j = 0; j < 49; ++j) {
        int rj = j / 7, cj = j % 7;
        int rel = (ri - rj + 6) * 13 + (ci - cj + 6);
        float sv = Si[j] * SCALE + rpb[rel * 6 + head];
        if (shift) {
          int regj = reg3(wr * 7 + rj) * 3 + reg3(wc * 7 + cj);
          if (regj != regi) sv -= 100.0f;
        }
        Si[j] = sv;
        mx = fmaxf(mx, sv);
      }
      float sum = 0.f;
      for (int j = 0; j < 49; ++j) {
        float e = __expf(Si[j] - mx);
        Si[j] = e; sum += e;
      }
      float rinv = 1.0f / sum;
      u16s* Pi = P + (size_t)i * 64;
      for (int j = 0; j < 49; ++j) Pi[j] = f2bf(Si[j] * rinv);  // in-place f32->bf16 (write offset <= read offset)
      for (int j = 49; j < 64; ++j) Pi[j] = 0;
    }
  }
  __syncthreads();

  // ---- out = P @ V (64x32, K=64 padded, 16 WMMAs) ----
  Frag bv[2][2];
#pragma unroll
  for (int nj = 0; nj < 2; ++nj)
#pragma unroll
    for (int kk = 0; kk < 2; ++kk)
#pragma unroll
      for (int v = 0; v < 8; ++v)
#pragma unroll
        for (int p = 0; p < 2; ++p) {
          int key = kk * 32 + v * 2 + p + grp * 16;
          bv[nj][kk].h[v * 2 + p] =
              (key < 49) ? qkv[base + (size_t)key * 576 + 384 + head * 32 + nj * 16 + lm]
                         : (u16s)0;
        }
#pragma unroll
  for (int mi = 0; mi < 4; ++mi) {
    int m = mi * 16 + lm;
    Frag a0, a1;
#pragma unroll
    for (int v = 0; v < 8; ++v) {
      int kb = v * 2 + grp * 16;
      a0.u[v] = (m < 49) ? *(const u32*)&P[(size_t)m * 64 + kb] : 0u;
      a1.u[v] = (m < 49) ? *(const u32*)&P[(size_t)m * 64 + 32 + kb] : 0u;
    }
#pragma unroll
    for (int nj = 0; nj < 2; ++nj) {
      v8f acc = {};
      acc = __builtin_amdgcn_wmma_f32_16x16x32_bf16(false, a0.v, false, bv[nj][0].v, (short)0, acc, false, false);
      acc = __builtin_amdgcn_wmma_f32_16x16x32_bf16(false, a1.v, false, bv[nj][1].v, (short)0, acc, false, false);
#pragma unroll
      for (int r = 0; r < 8; ++r) {
        int mo = mi * 16 + grp * 8 + r;
        if (mo < 49)
          out[((size_t)win * 49 + mo) * 192 + head * 32 + nj * 16 + lm] = f2bf(acc[r]);
      }
    }
  }
}

// ---------------------------------------------------------------- window-reverse + unshift + residual add
__global__ __launch_bounds__(256) void winrev_res_kernel(const float* __restrict__ resid,
                                                         const float* __restrict__ proj,
                                                         float* __restrict__ out, int shift) {
  size_t idx = (size_t)blockIdx.x * 256 + threadIdx.x;
  if (idx >= (size_t)64 * 3136 * 192) return;
  int c = (int)(idx % 192);
  size_t t = idx / 192;
  int j = (int)(t % 56);
  size_t t2 = t / 56;
  int i = (int)(t2 % 56);
  int bb = (int)(t2 / 56);
  int si = (i - shift + 56) % 56;  // roll back by +shift
  int sj = (j - shift + 56) % 56;
  int win = bb * 64 + (si / 7) * 8 + (sj / 7);
  int n = (si % 7) * 7 + (sj % 7);
  out[idx] = resid[idx] + proj[((size_t)win * 49 + n) * 192 + c];
}

// ---------------------------------------------------------------- host
extern "C" void kernel_launch(void* const* d_in, const int* in_sizes, int n_in,
                              void* d_out, int out_size, void* d_ws, size_t ws_size,
                              hipStream_t stream) {
  const float* x_in   = (const float*)d_in[0];
  const float* ln1_g  = (const float*)d_in[1];
  const float* ln1_b  = (const float*)d_in[2];
  const float* qkv_w  = (const float*)d_in[3];
  const float* qkv_b  = (const float*)d_in[4];
  const float* rpb    = (const float*)d_in[5];
  const float* proj_w = (const float*)d_in[6];
  const float* proj_b = (const float*)d_in[7];
  const float* ln2_g  = (const float*)d_in[8];
  const float* ln2_b  = (const float*)d_in[9];
  const float* fc1_w  = (const float*)d_in[10];
  const float* fc1_b  = (const float*)d_in[11];
  const float* fc2_w  = (const float*)d_in[12];
  const float* fc2_b  = (const float*)d_in[13];

  const int M = 200704;  // 4096 windows * 49 tokens
  char* ws = (char*)d_ws;
  size_t off = 0;
  auto take = [&](size_t bytes) -> char* {
    char* p = ws + off;
    off += (bytes + 255) & ~(size_t)255;
    return p;
  };
  float* xbuf   = (float*)take((size_t)M * 192 * 4);      // running residual x
  u16s*  bufA   = (u16s*) take((size_t)M * 192 * 2);      // LN-out / attn-out (bf16)
  u16s*  bufBig = (u16s*) take((size_t)M * 768 * 2);      // qkv (576) / fc1-out (768)
  float* bufF   = (float*)take((size_t)M * 192 * 4);      // proj-out (f32)
  u16s*  wqkv   = (u16s*) take((size_t)2 * 192 * 576 * 2);
  u16s*  wproj  = (u16s*) take((size_t)2 * 192 * 192 * 2);
  u16s*  wfc1   = (u16s*) take((size_t)2 * 192 * 768 * 2);
  u16s*  wfc2   = (u16s*) take((size_t)2 * 768 * 192 * 2);

  auto cvt = [&](const float* src, u16s* dst, int n) {
    cvt_bf16_kernel<<<(n + 255) / 256, 256, 0, stream>>>(src, dst, n);
  };
  cvt(qkv_w,  wqkv,  2 * 192 * 576);
  cvt(proj_w, wproj, 2 * 192 * 192);
  cvt(fc1_w,  wfc1,  2 * 192 * 768);
  cvt(fc2_w,  wfc2,  2 * 768 * 192);

  const int MT = M / 64;  // 3136 row tiles
  for (int d = 0; d < 2; ++d) {
    int shift = (d == 0) ? 0 : 3;
    const float* xsrc = (d == 0) ? x_in : xbuf;

    // LN1 + shift + window partition -> bf16
    ln_kernel<<<M / 8, 256, 0, stream>>>(xsrc, ln1_g + d * 192, ln1_b + d * 192, bufA, shift, 1);

    // QKV projection
    gemm_bf16_kernel<<<dim3(MT, 576 / 64), 256, 0, stream>>>(
        bufA, wqkv + (size_t)d * 192 * 576, qkv_b + d * 576, bufBig, nullptr, 192, 576, EPI_BF16);

    // fused windowed attention
    attn_kernel<<<4096, 192, 6 * 64 * 64 * 4, stream>>>(
        bufBig, rpb + (size_t)d * 169 * 6, bufA, shift);

    // output projection
    gemm_bf16_kernel<<<dim3(MT, 192 / 64), 256, 0, stream>>>(
        bufA, wproj + (size_t)d * 192 * 192, proj_b + d * 192, bufF, nullptr, 192, 192, EPI_F32);

    // window reverse + unshift + residual
    winrev_res_kernel<<<(int)(((size_t)M * 192 + 255) / 256), 256, 0, stream>>>(
        xsrc, bufF, xbuf, shift);

    // LN2 (plain token order)
    ln_kernel<<<M / 8, 256, 0, stream>>>(xbuf, ln2_g + d * 192, ln2_b + d * 192, bufA, 0, 0);

    // FC1 + exact GELU
    gemm_bf16_kernel<<<dim3(MT, 768 / 64), 256, 0, stream>>>(
        bufA, wfc1 + (size_t)d * 192 * 768, fc1_b + d * 768, bufBig, nullptr, 192, 768, EPI_GELU_BF16);

    // FC2 + residual; last block writes d_out directly
    float* outp = (d == 1) ? (float*)d_out : xbuf;
    gemm_bf16_kernel<<<dim3(MT, 192 / 64), 256, 0, stream>>>(
        bufBig, wfc2 + (size_t)d * 768 * 192, fc2_b + d * 192, outp, xbuf, 768, 192, EPI_RESID_F32);
  }
}
